// NcpBlock_64793876627652
// MI455X (gfx1250) — matrix operand order repaired and checked by reading, not compiled, not run on hardware
//
#include <hip/hip_runtime.h>
#include <stdint.h>

// ---------------------------------------------------------------------------
// Types / helpers
// ---------------------------------------------------------------------------
typedef __attribute__((ext_vector_type(16))) __bf16 v16bf;
typedef __attribute__((ext_vector_type(8)))  float  v8f;

union Frag { v16bf v; uint32_t u[8]; };

__device__ __forceinline__ unsigned short f2bf(float f) {
  union { float f; uint32_t u; } a; a.f = f;
  uint32_t u = a.u;
  uint32_t r = u + 0x7FFFu + ((u >> 16) & 1u);   // round-to-nearest-even
  return (unsigned short)(r >> 16);
}
__device__ __forceinline__ float bf2f(unsigned short h) {
  union { uint32_t u; float f; } a; a.u = ((uint32_t)h) << 16; return a.f;
}

// A fragment: 16x32 bf16 tile from row-major [16+, ld] at k = kt*32.
// Lane l -> row (l&15); lanes<16 hold K {0..7,16..23}, lanes>=16 hold {8..15,24..31}.
__device__ __forceinline__ void load_afrag(Frag& a, const unsigned short* src, int ld, int kt, int lane) {
  int r  = lane & 15;
  int ka = kt * 32 + ((lane >> 4) << 3);
  const unsigned short* p = src + r * ld + ka;
#pragma unroll
  for (int i = 0; i < 4; i++) {
    a.u[i]     = *(const uint32_t*)(p + 2 * i);
    a.u[4 + i] = *(const uint32_t*)(p + 16 + 2 * i);
  }
}

// B fragment: 32x16 bf16 tile (B = W^T) loaded from row-major W [N, ldw],
// rows nrow0..nrow0+15, k = kt*32. Lane l -> column (l&15);
// lanes<16 hold K {0..7,16..23}, lanes>=16 hold {8..15,24..31} (2 bf16 per VGPR).
__device__ __forceinline__ void load_bfrag(Frag& b, const unsigned short* w, int ldw, int nrow0, int kt, int lane) {
  int nr = nrow0 + (lane & 15);
  int kb = kt * 32 + ((lane >> 4) << 3);
  const unsigned short* p = w + (size_t)nr * ldw + kb;
#pragma unroll
  for (int i = 0; i < 4; i++) {
    b.u[i]     = *(const uint32_t*)(p + 2 * i);
    b.u[4 + i] = *(const uint32_t*)(p + 16 + 2 * i);
  }
}

// ---------------------------------------------------------------------------
// Kernel 1: repack weights to padded bf16 (mask pre-applied), biases padded f32
// Padded geometry:
//   layer0: hi 308->320, gates stride 320; W0X [1280][512], W0H [1280][320]
//   layer1: hc 204->208 (N) / 224 (K); W1 [832][544]  (K = 320 hi + 224 hc)
//   layer2: hm 512;                W2 [2048][736] (K = 224 hc + 512 hm)
// ---------------------------------------------------------------------------
struct WPtrs {
  const float* w[3][4];   // [layer][ff1,ff2,ta,tb]
  const float* b[3][4];
  const float* m[3];
};

__global__ __launch_bounds__(256) void prep_kernel(WPtrs P,
    unsigned short* __restrict__ w0x, unsigned short* __restrict__ w0h,
    unsigned short* __restrict__ w1p, unsigned short* __restrict__ w2p,
    float* __restrict__ b0p, float* __restrict__ b1p, float* __restrict__ b2p)
{
  const long S0 = 1280L * 512, S1 = 1280L * 320, S2 = 832L * 544, S3 = 2048L * 736;
  const long SB = 1280 + 832 + 2048;
  const long total = S0 + S1 + S2 + S3 + SB;
  for (long i = (long)blockIdx.x * 256 + threadIdx.x; i < total; i += (long)gridDim.x * 256) {
    long j = i;
    if (j < S0) {                                   // W0X: x part of layer0
      int r = (int)(j / 512), k = (int)(j % 512), g = r / 320, n = r % 320;
      float v = 0.f;
      if (n < 308) { v = P.w[0][g][n * 820 + k]; if (g < 2) v *= P.m[0][n * 820 + k]; }
      w0x[j] = f2bf(v);
    } else if ((j -= S0) < S1) {                    // W0H: recurrent part of layer0
      int r = (int)(j / 320), c = (int)(j % 320), g = r / 320, n = r % 320;
      float v = 0.f;
      if (n < 308 && c < 308) { v = P.w[0][g][n * 820 + 512 + c]; if (g < 2) v *= P.m[0][n * 820 + 512 + c]; }
      w0h[j] = f2bf(v);
    } else if ((j -= S1) < S2) {                    // W1
      int r = (int)(j / 544), c = (int)(j % 544), g = r / 208, n = r % 208;
      float v = 0.f;
      if (n < 204) {
        if (c < 320) { if (c < 308) { v = P.w[1][g][n * 512 + c]; if (g < 2) v *= P.m[1][n * 512 + c]; } }
        else { int k2 = c - 320; if (k2 < 204) { v = P.w[1][g][n * 512 + 308 + k2]; if (g < 2) v *= P.m[1][n * 512 + 308 + k2]; } }
      }
      w1p[j] = f2bf(v);
    } else if ((j -= S2) < S3) {                    // W2
      int r = (int)(j / 736), c = (int)(j % 736), g = r / 512, n = r % 512;
      float v = 0.f;
      if (c < 224) { if (c < 204) { v = P.w[2][g][n * 716 + c]; if (g < 2) v *= P.m[2][n * 716 + c]; } }
      else { int k2 = c - 224;      v = P.w[2][g][n * 716 + 204 + k2]; if (g < 2) v *= P.m[2][n * 716 + 204 + k2]; }
      w2p[j] = f2bf(v);
    } else {                                        // biases (padding -> 0)
      j -= S3;
      if (j < 1280)            { int g = (int)j / 320, n = (int)j % 320; b0p[j] = (n < 308) ? P.b[0][g][n] : 0.f; }
      else if ((j -= 1280) < 832) { int g = (int)j / 208, n = (int)j % 208; b1p[j] = (n < 204) ? P.b[1][g][n] : 0.f; }
      else { j -= 832;           int g = (int)j / 512, n = (int)j % 512; b2p[j] = P.b[2][g][n]; }
    }
  }
}

// ---------------------------------------------------------------------------
// Kernel 2: LayerNorm over E=512 -> bf16 xn [B*T, 512]
// ---------------------------------------------------------------------------
__global__ __launch_bounds__(256) void ln_kernel(const float* __restrict__ x,
    const float* __restrict__ gmm, const float* __restrict__ bta,
    unsigned short* __restrict__ xn)
{
  __shared__ float s1[8], s2[8];
  const int row = blockIdx.x;                 // row = b*T + t
  const float* xr = x + (size_t)row * 512;
  float v0 = xr[threadIdx.x], v1 = xr[threadIdx.x + 256];
  float s = v0 + v1, q = v0 * v0 + v1 * v1;
#pragma unroll
  for (int off = 16; off > 0; off >>= 1) {
    s += __shfl_xor(s, off, 32);
    q += __shfl_xor(q, off, 32);
  }
  int wv = threadIdx.x >> 5, ln = threadIdx.x & 31;
  if (ln == 0) { s1[wv] = s; s2[wv] = q; }
  __syncthreads();
  s = 0.f; q = 0.f;
#pragma unroll
  for (int i = 0; i < 8; i++) { s += s1[i]; q += s2[i]; }
  float mu  = s * (1.f / 512.f);
  float var = q * (1.f / 512.f) - mu * mu;
  float rs  = rsqrtf(var + 1e-5f);
  unsigned short* o = xn + (size_t)row * 512;
  o[threadIdx.x]       = f2bf((v0 - mu) * rs * gmm[threadIdx.x]       + bta[threadIdx.x]);
  o[threadIdx.x + 256] = f2bf((v1 - mu) * rs * gmm[threadIdx.x + 256] + bta[threadIdx.x + 256]);
}

// ---------------------------------------------------------------------------
// Kernel 3: Xp[t][b][1280] = xn @ W0X^T   (M=65536, N=1280, K=512) bf16 WMMA
// grid (4096, 10), block 256 = 8 waves; wave -> one 16x16 tile
// ---------------------------------------------------------------------------
__global__ __launch_bounds__(256) void xproj_kernel(const unsigned short* __restrict__ xn,
    const unsigned short* __restrict__ w0x, float* __restrict__ xp)
{
  const int lane = threadIdx.x & 31;
  const int wv   = threadIdx.x >> 5;
  const int mt   = blockIdx.x;                   // 0..4095 (row tiles of 16)
  const int nt   = blockIdx.y * 8 + wv;          // 0..79   (col tiles of 16)
  const unsigned short* abase = xn + (size_t)mt * 16 * 512;
  v8f acc = {0.f, 0.f, 0.f, 0.f, 0.f, 0.f, 0.f, 0.f};
  for (int kt = 0; kt < 16; ++kt) {
    Frag a, bq;
    load_afrag(a, abase, 512, kt, lane);
    load_bfrag(bq, w0x, 512, nt * 16, kt, lane);
    acc = __builtin_amdgcn_wmma_f32_16x16x32_bf16(false, a.v, false, bq.v, (short)0, acc, false, false);
  }
  const int n = lane & 15, mb = (lane >> 4) * 8;
#pragma unroll
  for (int v = 0; v < 8; ++v) {
    int gm = mt * 16 + mb + v;                   // global row = b*1024 + t
    int bb = gm >> 10, tt = gm & 1023;
    xp[((size_t)tt * 64 + bb) * 1280 + nt * 16 + n] = acc[v];
  }
}

// ---------------------------------------------------------------------------
// Kernel 4: persistent recurrence. 4 blocks x 16 waves; block owns 16 batch rows.
// ---------------------------------------------------------------------------
struct PhaseCfg {
  const unsigned short* a1; int ld1; int kt1;    // first K region (LDS)
  const unsigned short* a2; int ld2; int kt2;    // second K region (LDS) or null
  const unsigned short* w;  int ldw; int nstr;   // bf16 weights [4*nstr][ldw]
  const float* bias;                             // [4*nstr]
  const float* xinit;                            // layer0 only: row stride 1280, gate stride 320
  unsigned short* dst; int dld;                  // new h (bf16 LDS)
  float* yout; size_t ystride;                   // layer2 only
  int ntiles;                                    // N tiles (16 wide)
};

__device__ __forceinline__ void cell_phase(const PhaseCfg c, int wave, int lane) {
  const int n = lane & 15, mb = (lane >> 4) * 8;
  float res[2][8];
  int tids[2]; int tc = 0;
  for (int nt = wave; nt < c.ntiles; nt += 16) {
    v8f acc[4];
    if (c.xinit) {
#pragma unroll
      for (int g = 0; g < 4; g++) {
        const float* xb = c.xinit + g * 320 + nt * 16 + n;
#pragma unroll
        for (int v = 0; v < 8; v++) acc[g][v] = xb[(size_t)(mb + v) * 1280];
      }
    } else {
#pragma unroll
      for (int g = 0; g < 4; g++)
#pragma unroll
        for (int v = 0; v < 8; v++) acc[g][v] = 0.f;
    }
    const int ktot = c.kt1 + c.kt2;
    for (int kt = 0; kt < ktot; kt++) {
      Frag a;
      if (kt < c.kt1) load_afrag(a, c.a1, c.ld1, kt, lane);
      else            load_afrag(a, c.a2, c.ld2, kt - c.kt1, lane);
#pragma unroll
      for (int g = 0; g < 4; g++) {
        Frag bq;
        load_bfrag(bq, c.w, c.ldw, g * c.nstr + nt * 16, kt, lane);
        acc[g] = __builtin_amdgcn_wmma_f32_16x16x32_bf16(false, a.v, false, bq.v, (short)0, acc[g], false, false);
      }
    }
    const float bf1 = c.bias[0 * c.nstr + nt * 16 + n];
    const float bf2 = c.bias[1 * c.nstr + nt * 16 + n];
    const float bta = c.bias[2 * c.nstr + nt * 16 + n];
    const float btb = c.bias[3 * c.nstr + nt * 16 + n];
#pragma unroll
    for (int v = 0; v < 8; v++) {
      float f1 = tanhf(acc[0][v] + bf1);
      float f2 = tanhf(acc[1][v] + bf2);
      float ts = (acc[2][v] + bta) + (acc[3][v] + btb);
      float tt = 1.f / (1.f + __expf(-ts));
      res[tc][v] = f1 * (1.f - tt) + tt * f2;
    }
    tids[tc++] = nt;
  }
  __syncthreads();                                // all reads of old h done
  for (int i = 0; i < tc; i++) {
    int nt = tids[i];
#pragma unroll
    for (int v = 0; v < 8; v++) {
      float h = res[i][v];
      c.dst[(size_t)(mb + v) * c.dld + nt * 16 + n] = f2bf(h);
      if (c.yout) c.yout[(size_t)(mb + v) * c.ystride + nt * 16 + n] = h;
    }
  }
  __syncthreads();                                // new h visible
}

__global__ __launch_bounds__(512) void rnn_kernel(
    const float* __restrict__ xp,
    const unsigned short* __restrict__ w0h,
    const unsigned short* __restrict__ w1p,
    const unsigned short* __restrict__ w2p,
    const float* __restrict__ b0p, const float* __restrict__ b1p, const float* __restrict__ b2p,
    const float* __restrict__ h0,
    float* __restrict__ y, float* __restrict__ hfin)
{
  __shared__ __align__(16) unsigned short HI[16 * 320];
  __shared__ __align__(16) unsigned short HC[16 * 224];
  __shared__ __align__(16) unsigned short HM[16 * 512];
  const int tid = threadIdx.x;
  const int lane = tid & 31, wave = tid >> 5;
  const int brow = blockIdx.x * 16;

  for (int i = tid; i < 16 * 320; i += 512) HI[i] = 0;
  for (int i = tid; i < 16 * 224; i += 512) HC[i] = 0;
  for (int i = tid; i < 16 * 512; i += 512) HM[i] = 0;
  __syncthreads();
  for (int i = tid; i < 16 * 1024; i += 512) {
    int m = i >> 10, u = i & 1023;
    unsigned short bv = f2bf(h0[(size_t)(brow + m) * 1024 + u]);
    if (u < 308)      HI[m * 320 + u] = bv;
    else if (u < 512) HC[m * 224 + (u - 308)] = bv;
    else              HM[m * 512 + (u - 512)] = bv;
  }
  __syncthreads();

  for (int t = 0; t < 1024; ++t) {
    if (t + 1 < 1024) {                            // prefetch next step's Xp strip
      const float* nx = xp + ((size_t)(t + 1) * 64 + brow) * 1280;
      for (int i = tid * 32; i < 16 * 1280; i += 512 * 32)
        __builtin_prefetch(nx + i, 0, 1);
    }
    PhaseCfg c0 = { HI, 320, 10, nullptr, 0, 0,  w0h, 320, 320, b0p,
                    xp + ((size_t)t * 64 + brow) * 1280, HI, 320, nullptr, 0, 20 };
    cell_phase(c0, wave, lane);
    PhaseCfg c1 = { HI, 320, 10, HC, 224, 7,     w1p, 544, 208, b1p,
                    nullptr, HC, 224, nullptr, 0, 13 };
    cell_phase(c1, wave, lane);
    PhaseCfg c2 = { HC, 224, 7,  HM, 512, 16,    w2p, 736, 512, b2p,
                    nullptr, HM, 512,
                    y + (size_t)brow * 1024 * 512 + (size_t)t * 512, (size_t)1024 * 512, 32 };
    cell_phase(c2, wave, lane);
  }

  for (int i = tid; i < 16 * 1024; i += 512) {     // h_final [B,1024]
    int m = i >> 10, u = i & 1023;
    unsigned short bv;
    if (u < 308)      bv = HI[m * 320 + u];
    else if (u < 512) bv = HC[m * 224 + (u - 308)];
    else              bv = HM[m * 512 + (u - 512)];
    hfin[(size_t)(brow + m) * 1024 + u] = bf2f(bv);
  }
}

// ---------------------------------------------------------------------------
// Launch
// ---------------------------------------------------------------------------
extern "C" void kernel_launch(void* const* d_in, const int* in_sizes, int n_in,
                              void* d_out, int out_size, void* d_ws, size_t ws_size,
                              hipStream_t stream)
{
  (void)in_sizes; (void)n_in; (void)out_size; (void)ws_size;
  WPtrs P;
  int idx = 0;
  for (int l = 0; l < 3; ++l) {
    for (int j = 0; j < 4; ++j) {
      P.w[l][j] = (const float*)d_in[idx++];
      P.b[l][j] = (const float*)d_in[idx++];
    }
    P.m[l] = (const float*)d_in[idx++];
  }
  const float* x   = (const float*)d_in[27];
  const float* h0  = (const float*)d_in[28];
  const float* lng = (const float*)d_in[29];
  const float* lnb = (const float*)d_in[30];
  float* y    = (float*)d_out;
  float* hfin = y + (size_t)64 * 1024 * 512;

  char* p = (char*)d_ws;
  auto take = [&](size_t bytes) -> char* {
    char* q = p; p += (bytes + 255) & ~(size_t)255; return q;
  };
  unsigned short* xn  = (unsigned short*)take((size_t)64 * 1024 * 512 * 2);
  unsigned short* w0x = (unsigned short*)take((size_t)1280 * 512 * 2);
  unsigned short* w0h = (unsigned short*)take((size_t)1280 * 320 * 2);
  unsigned short* w1p = (unsigned short*)take((size_t)832 * 544 * 2);
  unsigned short* w2p = (unsigned short*)take((size_t)2048 * 736 * 2);
  float* b0p = (float*)take(1280 * 4);
  float* b1p = (float*)take(832 * 4);
  float* b2p = (float*)take(2048 * 4);
  float* xp  = (float*)take((size_t)1024 * 64 * 1280 * 4);

  hipLaunchKernelGGL(prep_kernel, dim3(2048), dim3(256), 0, stream,
                     P, w0x, w0h, w1p, w2p, b0p, b1p, b2p);
  hipLaunchKernelGGL(ln_kernel, dim3(64 * 1024), dim3(256), 0, stream, x, lng, lnb, xn);
  hipLaunchKernelGGL(xproj_kernel, dim3(4096, 10), dim3(256), 0, stream, xn, w0x, xp);
  hipLaunchKernelGGL(rnn_kernel, dim3(4), dim3(512), 0, stream,
                     xp, w0h, w1p, w2p, b0p, b1p, b2p, h0, y, hfin);
}
